// MoELayer_27238682591311
// MI455X (gfx1250) — matrix operand order repaired and checked by reading, not compiled, not run on hardware
//
#include <hip/hip_runtime.h>
#include <hip/hip_bf16.h>
#include <math.h>

// ---------------- problem constants ----------------
#define B_   4
#define S_   2048
#define D_   2048
#define E_   8
#define H_   5632
#define T_   (B_ * S_)          // 8192 tokens
#define TOPK 2
#define DPSL_PRIOR 0.125f
#define AUX_W 0.01f
#define SPR_W 0.1f
#define NSPR 256

typedef __attribute__((ext_vector_type(16))) __bf16 v16bf;
typedef __attribute__((ext_vector_type(8)))  __bf16 v8bf;
typedef __attribute__((ext_vector_type(4)))  __bf16 v4bf;
typedef __attribute__((ext_vector_type(2)))  __bf16 v2bf;
typedef __attribute__((ext_vector_type(8)))  float  v8f;

// ---------------- workspace layout (bytes) ----------------
#define WS_COMBINE   0
#define WS_SCALARS   (262144)
#define WS_PROBS256  (262144 + 256)
#define WS_XN        (WS_PROBS256 + NSPR * E_ * 4)
#define WS_RN        (WS_XN + NSPR * D_ * 4)
#define WS_ACT       (4u << 20)

// ================= router =================
__global__ void moe_router(const float* __restrict__ x,
                           const float* __restrict__ gate_w,
                           float* __restrict__ combine,
                           float* __restrict__ scalars,
                           float* __restrict__ probs256) {
  const int lane = threadIdx.x & 31;
  const int wave = threadIdx.x >> 5;
  const int t = blockIdx.x * 8 + wave;          // one wave per token
  const float* xr = x + (size_t)t * D_;

  float acc[E_];
#pragma unroll
  for (int e = 0; e < E_; ++e) acc[e] = 0.f;

  for (int d = lane; d < D_; d += 32) {
    float xv = xr[d];
#pragma unroll
    for (int e = 0; e < E_; ++e) acc[e] += xv * gate_w[(size_t)e * D_ + d];
  }
#pragma unroll
  for (int off = 16; off >= 1; off >>= 1) {
#pragma unroll
    for (int e = 0; e < E_; ++e) acc[e] += __shfl_xor(acc[e], off, 32);
  }

  if (lane == 0) {
    float m = acc[0];
#pragma unroll
    for (int e = 1; e < E_; ++e) m = fmaxf(m, acc[e]);
    float p[E_], sum = 0.f;
#pragma unroll
    for (int e = 0; e < E_; ++e) { p[e] = __expf(acc[e] - m); sum += p[e]; }
    float inv = 1.f / sum;
#pragma unroll
    for (int e = 0; e < E_; ++e) p[e] *= inv;

    int i1 = 0;
#pragma unroll
    for (int e = 1; e < E_; ++e) if (p[e] > p[i1]) i1 = e;
    int i2 = (i1 == 0) ? 1 : 0;
#pragma unroll
    for (int e = 0; e < E_; ++e) if (e != i1 && p[e] > p[i2]) i2 = e;

    float wsum = p[i1] + p[i2];
    float w1v = p[i1] / wsum, w2v = p[i2] / wsum;
#pragma unroll
    for (int e = 0; e < E_; ++e)
      combine[(size_t)t * E_ + e] = (e == i1) ? w1v : ((e == i2) ? w2v : 0.f);

#pragma unroll
    for (int e = 0; e < E_; ++e) atomicAdd(&scalars[e], p[e]);   // P_i sums
    atomicAdd(&scalars[8 + i1], 1.f);                            // expert counts
    atomicAdd(&scalars[8 + i2], 1.f);

    if (t < NSPR) {
#pragma unroll
      for (int e = 0; e < E_; ++e) probs256[t * E_ + e] = p[e];
    }
  }
}

// ================= SPR loss helpers =================
__global__ void init_scalars(float* scalars) {
  if (threadIdx.x < 32) scalars[threadIdx.x] = 0.f;
}

__global__ void spr_prep(const float* __restrict__ x,
                         const float* __restrict__ probs256,
                         float* __restrict__ xn, float* __restrict__ rn) {
  const int i = blockIdx.x;                      // token 0..255
  __shared__ float red[256];
  float s = 0.f;
  for (int d = threadIdx.x; d < D_; d += 256) {
    float v = x[(size_t)i * D_ + d]; s += v * v;
  }
  red[threadIdx.x] = s; __syncthreads();
  for (int st = 128; st > 0; st >>= 1) {
    if (threadIdx.x < st) red[threadIdx.x] += red[threadIdx.x + st];
    __syncthreads();
  }
  float inv = 1.f / fmaxf(sqrtf(red[0]), 1e-12f);
  for (int d = threadIdx.x; d < D_; d += 256)
    xn[(size_t)i * D_ + d] = x[(size_t)i * D_ + d] * inv;
  if (threadIdx.x == 0) {
    float s2 = 0.f;
#pragma unroll
    for (int e = 0; e < E_; ++e) { float p = probs256[i * E_ + e]; s2 += p * p; }
    float inv2 = 1.f / fmaxf(sqrtf(s2), 1e-12f);
#pragma unroll
    for (int e = 0; e < E_; ++e) rn[i * E_ + e] = probs256[i * E_ + e] * inv2;
  }
}

__global__ void spr_gram(const float* __restrict__ xn,
                         const float* __restrict__ rn,
                         float* __restrict__ scalars) {
  const int j = blockIdx.x;
  __shared__ float xj[D_];
  for (int d = threadIdx.x; d < D_; d += 256) xj[d] = xn[(size_t)j * D_ + d];
  __syncthreads();
  const int i = threadIdx.x;
  float dx = 0.f;
  for (int d = 0; d < D_; ++d) dx += xj[d] * xn[(size_t)i * D_ + d];
  float dr = 0.f;
#pragma unroll
  for (int e = 0; e < E_; ++e) dr += rn[j * E_ + e] * rn[i * E_ + e];
  float diff = dr - dx;
  __shared__ float red[256];
  red[threadIdx.x] = diff * diff; __syncthreads();
  for (int st = 128; st > 0; st >>= 1) {
    if (threadIdx.x < st) red[threadIdx.x] += red[threadIdx.x + st];
    __syncthreads();
  }
  if (threadIdx.x == 0) atomicAdd(&scalars[16], red[0]);
}

__global__ void finalize_aux(const float* __restrict__ scalars,
                             float* __restrict__ aux_out) {
  if (threadIdx.x == 0) {
    float lb = 0.f, dpsl = 0.f;
#pragma unroll
    for (int e = 0; e < E_; ++e) {
      float P = scalars[e] / (float)T_;
      float f = scalars[8 + e] / (float)(T_ * TOPK);
      lb += f * P;
      dpsl += DPSL_PRIOR * (__logf(DPSL_PRIOR) - __logf(fmaxf(P, 1e-30f)));
    }
    lb *= (float)E_;
    float spr = scalars[16] / (float)(NSPR * NSPR);
    aux_out[0] = AUX_W * (lb + dpsl + SPR_W * spr);
  }
}

// ================= fragment helpers =================
// Build a 16-half fragment from two aligned 8-half LDS runs (2x ds_load_b128).
__device__ __forceinline__ v16bf ld2(const __bf16* p0, const __bf16* p1) {
  v8bf lo = *(const v8bf*)p0;
  v8bf hi = *(const v8bf*)p1;
  return __builtin_shufflevector(lo, hi, 0, 1, 2, 3, 4, 5, 6, 7,
                                 8, 9, 10, 11, 12, 13, 14, 15);
}

#define LDA2 40   // A row stride in halves (80 B: 16-B aligned, skewed banks)
#define LDBT 40   // B^T col stride in halves

// A-frag 16x32 bf16 (ISA 7.12.2): lane<16 row=lane holds K 0..7,16..23; lane>=16 K 8..15,24..31
__device__ __forceinline__ v16bf load_a_frag(const __bf16* Arow0, int lane) {
  const int row = lane & 15;
  const int klo = (lane < 16) ? 0 : 8;
  const __bf16* p = Arow0 + row * LDA2 + klo;
  return ld2(p, p + 16);
}
// B-frag 32x16 bf16 from transposed LDS tile Bt[col][k]: lane<16 col=lane K 0..7,16..23; lane>=16 K 8..15,24..31
__device__ __forceinline__ v16bf load_b_frag_t(const __bf16* Btcol0, int lane) {
  const int col = lane & 15;
  const int kb0 = (lane < 16) ? 0 : 8;
  const __bf16* p = Btcol0 + col * LDBT + kb0;
  return ld2(p, p + 16);
}

// ================= GEMM A: act = silu(x@W1) * (x@W2)  (per expert) =================
// block tile 128(M=T) x 64(N=H), 8 waves in 4x2, per-wave 32x32 (2x2 WMMA tiles)
__global__ void __launch_bounds__(256)
moe_ffn_gate(const float* __restrict__ x,
             const float* __restrict__ w1, const float* __restrict__ w2,
             __bf16* __restrict__ act, int e) {
  __shared__ __bf16 As[128][LDA2];
  __shared__ __bf16 B1t[64][LDBT];   // transposed: [n][k]
  __shared__ __bf16 B2t[64][LDBT];

  const int tid = threadIdx.x, lane = tid & 31, wave = tid >> 5;
  const int wm = wave >> 1, wn = wave & 1;
  const int m0 = blockIdx.y * 128;
  const int n0 = blockIdx.x * 64;
  const float* w1e = w1 + (size_t)e * D_ * H_;
  const float* w2e = w2 + (size_t)e * D_ * H_;

  const v8f vz = {0.f, 0.f, 0.f, 0.f, 0.f, 0.f, 0.f, 0.f};
  v8f acc1[2][2], acc2[2][2];
#pragma unroll
  for (int i = 0; i < 2; ++i)
#pragma unroll
    for (int j = 0; j < 2; ++j) { acc1[i][j] = vz; acc2[i][j] = vz; }

  const int wr = (tid >> 4) << 1;   // weight sub-tile: rows wr, wr+1 (K dim)
  const int wc = (tid & 15) << 2;   // cols wc..wc+3 (N dim)

  for (int kb = 0; kb < D_; kb += 32) {
    if (kb + 32 < D_) {   // stream-prefetch next weight rows -> global_prefetch_b8
      __builtin_prefetch(&w1e[(size_t)(kb + 32) * H_ + n0], 0, 0);
      __builtin_prefetch(&w2e[(size_t)(kb + 32) * H_ + n0], 0, 0);
    }
    // x tile: 128x32 f32 -> bf16, b128 global loads, b64 LDS stores
#pragma unroll
    for (int i = 0; i < 4; ++i) {
      int idx = tid + i * 256;
      int m = idx >> 3, k4 = (idx & 7) << 2;
      const float4 v = *(const float4*)&x[(size_t)(m0 + m) * D_ + kb + k4];
      v4bf bv = {(__bf16)v.x, (__bf16)v.y, (__bf16)v.z, (__bf16)v.w};
      *(v4bf*)&As[m][k4] = bv;
    }
    // weight tiles: each thread loads a 2(K)x4(N) f32 sub-tile (2x b128),
    // stores transposed as 4x packed-pair b32 into B^T
    {
      size_t base = (size_t)(kb + wr) * H_ + n0 + wc;
      float4 u0 = *(const float4*)&w1e[base];
      float4 u1 = *(const float4*)&w1e[base + H_];
      float4 q0 = *(const float4*)&w2e[base];
      float4 q1 = *(const float4*)&w2e[base + H_];
      const float* pu0 = (const float*)&u0;
      const float* pu1 = (const float*)&u1;
      const float* pq0 = (const float*)&q0;
      const float* pq1 = (const float*)&q1;
#pragma unroll
      for (int j = 0; j < 4; ++j) {
        v2bf t1 = {(__bf16)pu0[j], (__bf16)pu1[j]};
        v2bf t2 = {(__bf16)pq0[j], (__bf16)pq1[j]};
        *(v2bf*)&B1t[wc + j][wr] = t1;
        *(v2bf*)&B2t[wc + j][wr] = t2;
      }
    }
    __syncthreads();

    v16bf a[2], b1[2], b2[2];
#pragma unroll
    for (int tm = 0; tm < 2; ++tm)
      a[tm] = load_a_frag(&As[wm * 32 + tm * 16][0], lane);
#pragma unroll
    for (int tn = 0; tn < 2; ++tn) {
      b1[tn] = load_b_frag_t(&B1t[wn * 32 + tn * 16][0], lane);
      b2[tn] = load_b_frag_t(&B2t[wn * 32 + tn * 16][0], lane);
    }
#pragma unroll
    for (int tm = 0; tm < 2; ++tm)
#pragma unroll
      for (int tn = 0; tn < 2; ++tn) {
        acc1[tm][tn] = __builtin_amdgcn_wmma_f32_16x16x32_bf16(
            false, a[tm], false, b1[tn], (short)0, acc1[tm][tn], false, false);
        acc2[tm][tn] = __builtin_amdgcn_wmma_f32_16x16x32_bf16(
            false, a[tm], false, b2[tn], (short)0, acc2[tm][tn], false, false);
      }
    __syncthreads();
  }

  // epilogue: act = silu(s1) * s2 (fast rcp), stored bf16
  const int col = lane & 15, r0 = (lane < 16) ? 0 : 8;
#pragma unroll
  for (int tm = 0; tm < 2; ++tm)
#pragma unroll
    for (int tn = 0; tn < 2; ++tn)
#pragma unroll
      for (int r = 0; r < 8; ++r) {
        float s1 = acc1[tm][tn][r], s2 = acc2[tm][tn][r];
        float sig = __builtin_amdgcn_rcpf(1.f + __expf(-s1));
        float v = s1 * sig * s2;
        int m = m0 + wm * 32 + tm * 16 + r0 + r;
        int h = n0 + wn * 32 + tn * 16 + col;
        act[(size_t)m * H_ + h] = (__bf16)v;
      }
}

// ================= GEMM B: out (+)= combine[:,e] * (act @ W3)  (per expert) =================
__global__ void __launch_bounds__(256)
moe_ffn_down(const __bf16* __restrict__ act,
             const float* __restrict__ w3,
             const float* __restrict__ combine,
             float* __restrict__ out, int e) {
  __shared__ __bf16 As[128][LDA2];
  __shared__ __bf16 Bt[64][LDBT];    // transposed: [n][k]

  const int tid = threadIdx.x, lane = tid & 31, wave = tid >> 5;
  const int wm = wave >> 1, wn = wave & 1;
  const int m0 = blockIdx.y * 128;
  const int n0 = blockIdx.x * 64;
  const float* w3e = w3 + (size_t)e * H_ * D_;

  const v8f vz = {0.f, 0.f, 0.f, 0.f, 0.f, 0.f, 0.f, 0.f};
  v8f acc[2][2];
#pragma unroll
  for (int i = 0; i < 2; ++i)
#pragma unroll
    for (int j = 0; j < 2; ++j) acc[i][j] = vz;

  const int wr = (tid >> 4) << 1;
  const int wc = (tid & 15) << 2;

  for (int kb = 0; kb < H_; kb += 32) {
    if (kb + 32 < H_)
      __builtin_prefetch(&w3e[(size_t)(kb + 32) * D_ + n0], 0, 0);
    // act tile 128x32 bf16: b128 global loads, b128 LDS stores
#pragma unroll
    for (int i = 0; i < 2; ++i) {
      int idx = tid + i * 256;
      int m = idx >> 2, k8 = (idx & 3) << 3;
      v8bf av = *(const v8bf*)&act[(size_t)(m0 + m) * H_ + kb + k8];
      *(v8bf*)&As[m][k8] = av;
    }
    // W3 tile: 2(K)x4(N) f32 sub-tile per thread -> transposed bf16 pairs
    {
      size_t base = (size_t)(kb + wr) * D_ + n0 + wc;
      float4 u0 = *(const float4*)&w3e[base];
      float4 u1 = *(const float4*)&w3e[base + D_];
      const float* pu0 = (const float*)&u0;
      const float* pu1 = (const float*)&u1;
#pragma unroll
      for (int j = 0; j < 4; ++j) {
        v2bf t = {(__bf16)pu0[j], (__bf16)pu1[j]};
        *(v2bf*)&Bt[wc + j][wr] = t;
      }
    }
    __syncthreads();

    v16bf a[2], b[2];
#pragma unroll
    for (int tm = 0; tm < 2; ++tm)
      a[tm] = load_a_frag(&As[wm * 32 + tm * 16][0], lane);
#pragma unroll
    for (int tn = 0; tn < 2; ++tn)
      b[tn] = load_b_frag_t(&Bt[wn * 32 + tn * 16][0], lane);
#pragma unroll
    for (int tm = 0; tm < 2; ++tm)
#pragma unroll
      for (int tn = 0; tn < 2; ++tn)
        acc[tm][tn] = __builtin_amdgcn_wmma_f32_16x16x32_bf16(
            false, a[tm], false, b[tn], (short)0, acc[tm][tn], false, false);
    __syncthreads();
  }

  const int col = lane & 15, r0 = (lane < 16) ? 0 : 8;
#pragma unroll
  for (int tm = 0; tm < 2; ++tm)
#pragma unroll
    for (int r = 0; r < 8; ++r) {
      int m = m0 + wm * 32 + tm * 16 + r0 + r;
      float cw = combine[(size_t)m * E_ + e];
#pragma unroll
      for (int tn = 0; tn < 2; ++tn) {
        int d = n0 + wn * 32 + tn * 16 + col;
        size_t o = (size_t)m * D_ + d;
        float v = cw * acc[tm][tn][r];
        if (e == 0) out[o] = v;      // beta = 0 on first expert (d_out is poisoned)
        else        out[o] += v;
      }
    }
}

// ================= host launch =================
extern "C" void kernel_launch(void* const* d_in, const int* in_sizes, int n_in,
                              void* d_out, int out_size, void* d_ws, size_t ws_size,
                              hipStream_t stream) {
  (void)in_sizes; (void)n_in; (void)out_size; (void)ws_size;
  const float* x      = (const float*)d_in[0];
  const float* gate_w = (const float*)d_in[1];
  const float* w1     = (const float*)d_in[2];
  const float* w2     = (const float*)d_in[3];
  const float* w3     = (const float*)d_in[4];
  float* out = (float*)d_out;

  char* ws = (char*)d_ws;
  float*  combine  = (float*)(ws + WS_COMBINE);
  float*  scalars  = (float*)(ws + WS_SCALARS);
  float*  probs256 = (float*)(ws + WS_PROBS256);
  float*  xn       = (float*)(ws + WS_XN);
  float*  rn       = (float*)(ws + WS_RN);
  __bf16* act      = (__bf16*)(ws + WS_ACT);

  init_scalars<<<1, 32, 0, stream>>>(scalars);
  moe_router<<<T_ / 8, 256, 0, stream>>>(x, gate_w, combine, scalars, probs256);
  spr_prep<<<NSPR, 256, 0, stream>>>(x, probs256, xn, rn);
  spr_gram<<<NSPR, 256, 0, stream>>>(xn, rn, scalars);

  for (int e = 0; e < E_; ++e) {
    moe_ffn_gate<<<dim3(H_ / 64, T_ / 128), 256, 0, stream>>>(x, w1, w2, act, e);
    moe_ffn_down<<<dim3(D_ / 64, T_ / 128), 256, 0, stream>>>(act, w3, combine, out, e);
  }
  finalize_aux<<<1, 32, 0, stream>>>(scalars, out + (size_t)T_ * D_);
}
